// hetero_full_info_actor_54193897341217
// MI455X (gfx1250) — matrix operand order repaired and checked by reading, not compiled, not run on hardware
//
#include <hip/hip_runtime.h>
#include <hip/hip_bf16.h>
#include <math.h>

#define D 128

typedef __attribute__((ext_vector_type(16))) __bf16 v16bf;
typedef __attribute__((ext_vector_type(8)))  float  v8f;

#if defined(__AMDGCN__) && __has_builtin(__builtin_amdgcn_global_load_async_to_lds_b128) && __has_builtin(__builtin_amdgcn_s_wait_asynccnt)
#define USE_ASYNC 1
typedef __attribute__((__vector_size__(4 * sizeof(int)))) int vint4;     // native vector, not HIP_vector_type
typedef __attribute__((address_space(1))) vint4 as1_vint4;
typedef __attribute__((address_space(3))) vint4 as3_vint4;
#else
#define USE_ASYNC 0
#endif

// ---------------- zero fill ----------------
__global__ void k_zero(float* __restrict__ p, int n) {
  int i = blockIdx.x * blockDim.x + threadIdx.x;
  int stride = gridDim.x * blockDim.x;
  for (; i < n; i += stride) p[i] = 0.0f;
}

// ---------------- f32 -> bf16 convert (8 elements / thread / iter) ----------------
__global__ void k_cvt_bf16(const float* __restrict__ in, __bf16* __restrict__ out, int n8) {
  int i = blockIdx.x * blockDim.x + threadIdx.x;
  int stride = gridDim.x * blockDim.x;
  for (; i < n8; i += stride) {
    const float4 a = *(const float4*)(in + (size_t)i * 8);
    const float4 b = *(const float4*)(in + (size_t)i * 8 + 4);
    union { __bf16 h[8]; uint4 u; } pk;
    pk.h[0] = (__bf16)a.x; pk.h[1] = (__bf16)a.y;
    pk.h[2] = (__bf16)a.z; pk.h[3] = (__bf16)a.w;
    pk.h[4] = (__bf16)b.x; pk.h[5] = (__bf16)b.y;
    pk.h[6] = (__bf16)b.z; pk.h[7] = (__bf16)b.w;
    *(uint4*)(out + (size_t)i * 8) = pk.u;
  }
}

// ---------------- embed: h = x @ W + b ; x:(N,36), W:(36,128) -> bf16 ----------------
__global__ void __launch_bounds__(128) k_embed(const float* __restrict__ x,
                                               const float* __restrict__ W,
                                               const float* __restrict__ b,
                                               __bf16* __restrict__ h) {
  __shared__ float xs[36];
  const int node = blockIdx.x;
  const int c = threadIdx.x;                 // 0..127
  if (c < 36) xs[c] = x[(size_t)node * 36 + c];
  __syncthreads();
  float acc = b[c];
#pragma unroll
  for (int k = 0; k < 36; ++k) acc = fmaf(xs[k], W[k * D + c], acc);
  h[(size_t)node * D + c] = (__bf16)acc;
}

// ---------------- edge scatter-add: bf16 gather -> f32 atomic agg ----------------
__global__ void k_scatter(const __bf16* __restrict__ h, const int* __restrict__ src,
                          const int* __restrict__ dst, float* __restrict__ agg, int E) {
  int gid = blockIdx.x * blockDim.x + threadIdx.x;
  if (gid >= E * 16) return;                 // 16 threads per edge, 8 ch each
  int e = gid >> 4, c8 = (gid & 15) << 3;
  union { uint4 u; __bf16 b[8]; } v;
  v.u = *(const uint4*)(h + (size_t)src[e] * D + c8);
  float* a = agg + (size_t)dst[e] * D + c8;
#pragma unroll
  for (int i = 0; i < 8; ++i) atomicAdd(a + i, (float)v.b[i]);
}

// ---------------- weight pack: f32 (128x128 row-major, W[k][n]) ->
//   bf16 in B-fragment lane order for v_wmma_f32_16x16x32_bf16 ----------
__global__ void __launch_bounds__(256) k_packw(const float* __restrict__ W,
                                               __bf16* __restrict__ dstbase, int ktBase) {
  int gid = blockIdx.x * blockDim.x + threadIdx.x;   // 16384 threads per matrix
  int j    = gid & 15;
  int lane = (gid >> 4) & 31;
  int nt   = (gid >> 9) & 7;
  int ktl  = gid >> 12;                              // 0..3
  int k = ktl * 32 + ((lane >= 16) ? 16 : 0) + j;
  int n = nt * 16 + (lane & 15);
  int kt = ktBase + ktl;
  dstbase[(((kt * 8 + nt) * 32 + lane) << 4) + j] = (__bf16)W[k * D + n];
}

// ---------------- fused multi-segment GEMM + bias + tanh (WMMA bf16) ----------------
// out(M,128) = tanh( sum_s A_s(M,128) @ W_s(128,128) + bias1 (+ bias2) ), all A bf16.
// 256 thr / 8 waves; M-tile 64 rows, N=128. Wave w: N-tile w, 4 M-subtiles.
// A tiles double-buffered in LDS; staged via GLOBAL_LOAD_ASYNC_TO_LDS (ASYNCcnt),
// decoupled from the B-fragment LOADcnt waits; one barrier per K-step.
template <int NSEG>
__global__ void __launch_bounds__(256)
k_gemm_tanh(const __bf16* __restrict__ a0, const __bf16* __restrict__ a1,
            const __bf16* __restrict__ a2, const __bf16* __restrict__ a3,
            const __bf16* __restrict__ wp,
            const float* __restrict__ bias1, const float* __restrict__ bias2,
            __bf16* __restrict__ out) {
  // 2 x (64 rows x 32 k bf16, pitch 40 bf16 = 80 B) = 10 KB
  __shared__ alignas(16) __bf16 atile[2 * 64 * 40];
  const int tid = threadIdx.x, lane = tid & 31, wave = tid >> 5;
  const int m0 = blockIdx.x * 64;
  const int srow = lane & 15, sel = lane >> 4;
  const int st_row = tid >> 2;               // 0..63
  const int st_k = (tid & 3) << 3;           // 0,8,16,24

  v8f acc[4];
#pragma unroll
  for (int mt = 0; mt < 4; ++mt)
#pragma unroll
    for (int i = 0; i < 8; ++i) acc[mt][i] = 0.0f;

  const __bf16* const seg[4] = {a0, a1, a2, a3};
  constexpr int KTN = NSEG * 4;

  // stage this thread's 16 B of tile kt into LDS buffer (kt&1)
  auto stage = [&](int kt) {
    const __bf16* src = seg[kt >> 2] + (size_t)(m0 + st_row) * D + (kt & 3) * 32 + st_k;
    char* dst = (char*)atile + (kt & 1) * 5120 + st_row * 80 + st_k * 2;
#if USE_ASYNC
    void* sv = const_cast<void*>(static_cast<const void*>(src));
    __builtin_amdgcn_global_load_async_to_lds_b128(
        (as1_vint4*)sv, (as3_vint4*)(void*)dst, 0, 0);
#else
    *(uint4*)dst = *(const uint4*)src;
#endif
  };

  stage(0);                                  // prologue
#pragma unroll
  for (int kt = 0; kt < KTN; ++kt) {
#if USE_ASYNC
    __builtin_amdgcn_s_wait_asynccnt(0);     // my writes of tile kt done
#endif
    __syncthreads();                         // everyone's tile-kt writes visible;
                                             // all waves done reading tile kt-1
    if (kt + 1 < KTN) stage(kt + 1);         // overlaps with WMMAs below

    // B fragment: contiguous 32 B per lane from packed weights (L2-resident)
    union { uint4 u[2]; v16bf v; } bfr;
    const __bf16* bb = wp + ((((size_t)kt * 8 + wave) * 32 + lane) << 4);
    bfr.u[0] = *(const uint4*)(bb);
    bfr.u[1] = *(const uint4*)(bb + 8);

    const char* buf = (const char*)atile + (kt & 1) * 5120;
#pragma unroll
    for (int mt = 0; mt < 4; ++mt) {
      union { uint4 u[2]; v16bf v; } af;
      const char* ab = buf + (mt * 16 + srow) * 80 + sel * 32;
      af.u[0] = *(const uint4*)(ab);
      af.u[1] = *(const uint4*)(ab + 16);
      acc[mt] = __builtin_amdgcn_wmma_f32_16x16x32_bf16(
          false, af.v, false, bfr.v, (short)0, acc[mt], false, false);
    }
  }

  // ---- epilogue: bias + tanh, bf16 store
  const int col = (wave << 4) + (lane & 15);
  float b = bias1[col];
  if (bias2) b += bias2[col];
#pragma unroll
  for (int mt = 0; mt < 4; ++mt) {
#pragma unroll
    for (int g = 0; g < 8; ++g) {
      int r = m0 + mt * 16 + g + sel * 8;
      out[(size_t)r * D + col] = (__bf16)tanhf(acc[mt][g] + b);
    }
  }
}

// ---------------- per-joint heads: loc / biased-softplus scale ----------------
__global__ void __launch_bounds__(256)
k_heads(const __bf16* __restrict__ h, const float* __restrict__ Wout,
        const float* __restrict__ bout, float* __restrict__ loc,
        float* __restrict__ scale, int NJ) {
  const int wave = threadIdx.x >> 5, lane = threadIdx.x & 31;
  const int n = blockIdx.x * 8 + wave;
  if (n >= NJ) return;
  const int j = n & 7;
  float a0 = 0.f, a1 = 0.f;
#pragma unroll
  for (int t = 0; t < 4; ++t) {
    int cidx = lane + t * 32;
    float hv = (float)h[(size_t)n * D + cidx];
    a0 = fmaf(hv, Wout[(j * D + cidx) * 2 + 0], a0);
    a1 = fmaf(hv, Wout[(j * D + cidx) * 2 + 1], a1);
  }
#pragma unroll
  for (int off = 16; off > 0; off >>= 1) {
    a0 += __shfl_down(a0, off, 32);
    a1 += __shfl_down(a1, off, 32);
  }
  if (lane == 0) {
    loc[n] = a0 + bout[j * 2 + 0];
    float t = a1 + bout[j * 2 + 1] + 0.5413248546129181f;  // log(e-1)
    float sp = (t > 20.0f) ? t : log1pf(expf(t));
    scale[n] = fmaxf(sp, 1e-4f);
  }
}

extern "C" void kernel_launch(void* const* d_in, const int* in_sizes, int n_in,
                              void* d_out, int out_size, void* d_ws, size_t ws_size,
                              hipStream_t stream) {
  const float* x_joint   = (const float*)d_in[0];
  const float* x_torso   = (const float*)d_in[1];
  const float* W_joint   = (const float*)d_in[2];
  const float* b_joint   = (const float*)d_in[3];
  const float* W_torso   = (const float*)d_in[4];
  const float* b_torso   = (const float*)d_in[5];
  const float* Wrel_thj1 = (const float*)d_in[6];
  const float* brel_thj1 = (const float*)d_in[7];
  const float* Wroot_thj1= (const float*)d_in[8];
  const float* Wrel_jht1 = (const float*)d_in[9];
  const float* brel_jht1 = (const float*)d_in[10];
  const float* Wroot_jht1= (const float*)d_in[11];
  const float* Wrel_jkj1 = (const float*)d_in[12];
  const float* brel_jkj1 = (const float*)d_in[13];
  const float* Wroot_jkj1= (const float*)d_in[14];
  const float* Wrel_thj2 = (const float*)d_in[15];
  const float* brel_thj2 = (const float*)d_in[16];
  const float* Wroot_thj2= (const float*)d_in[17];
  // d_in[18..20]: torso layer-2 weights, unused by the reference forward
  const float* Wrel_jkj2 = (const float*)d_in[21];
  const float* brel_jkj2 = (const float*)d_in[22];
  const float* Wroot_jkj2= (const float*)d_in[23];
  const float* W_out     = (const float*)d_in[24];
  const float* b_out     = (const float*)d_in[25];
  const int* thj_src = (const int*)d_in[26];
  const int* thj_dst = (const int*)d_in[27];
  const int* jht_src = (const int*)d_in[28];
  const int* jht_dst = (const int*)d_in[29];
  const int* jkj_src = (const int*)d_in[30];
  const int* jkj_dst = (const int*)d_in[31];

  const int NJ = in_sizes[0] / 36;
  const int NT = in_sizes[1] / 36;
  const int E1 = in_sizes[26], E2 = in_sizes[28], E3 = in_sizes[30];

  // -------- workspace carve-out --------
  char* ws = (char*)d_ws;
  size_t off = 0;
  auto carve = [&](size_t bytes) {
    char* p = ws + off;
    off += (bytes + 511) & ~(size_t)511;
    return p;
  };
  __bf16* hjb   = (__bf16*)carve((size_t)NJ * D * 2);  // embed out / reused as hj2
  __bf16* htb   = (__bf16*)carve((size_t)NT * D * 2);
  float*  aggA  = (float*)carve((size_t)NJ * D * 4);   // f32 atomic accum (thj)
  float*  aggB  = (float*)carve((size_t)NJ * D * 4);   // f32 atomic accum (jkj)
  float*  aggT  = (float*)carve((size_t)NT * D * 4);   // f32 atomic accum (jht)
  __bf16* aggAb = (__bf16*)carve((size_t)NJ * D * 2);  // bf16 copies for GEMM A
  __bf16* aggBb = (__bf16*)carve((size_t)NJ * D * 2);
  __bf16* aggTb = (__bf16*)carve((size_t)NT * D * 2);
  __bf16* hj1b  = (__bf16*)carve((size_t)NJ * D * 2);
  __bf16* ht1b  = (__bf16*)carve((size_t)NT * D * 2);
  __bf16* pw1 = (__bf16*)carve((size_t)4 * 16384 * 2);
  __bf16* pwT = (__bf16*)carve((size_t)2 * 16384 * 2);
  __bf16* pw2 = (__bf16*)carve((size_t)4 * 16384 * 2);
  __bf16* hj2b = hjb;   // hjb dead after layer-1 -> reuse

  // -------- weight packing (tiny) --------
  k_packw<<<64, 256, 0, stream>>>(Wrel_thj1,  pw1, 0);
  k_packw<<<64, 256, 0, stream>>>(Wroot_thj1, pw1, 4);
  k_packw<<<64, 256, 0, stream>>>(Wrel_jkj1,  pw1, 8);
  k_packw<<<64, 256, 0, stream>>>(Wroot_jkj1, pw1, 12);
  k_packw<<<64, 256, 0, stream>>>(Wrel_jht1,  pwT, 0);
  k_packw<<<64, 256, 0, stream>>>(Wroot_jht1, pwT, 4);
  k_packw<<<64, 256, 0, stream>>>(Wrel_thj2,  pw2, 0);
  k_packw<<<64, 256, 0, stream>>>(Wroot_thj2, pw2, 4);
  k_packw<<<64, 256, 0, stream>>>(Wrel_jkj2,  pw2, 8);
  k_packw<<<64, 256, 0, stream>>>(Wroot_jkj2, pw2, 12);

  // -------- input embed (bf16 out) --------
  k_embed<<<NJ, 128, 0, stream>>>(x_joint, W_joint, b_joint, hjb);
  k_embed<<<NT, 128, 0, stream>>>(x_torso, W_torso, b_torso, htb);

  // -------- layer 1 aggregations --------
  k_zero<<<4096, 256, 0, stream>>>(aggA, NJ * D);
  k_zero<<<4096, 256, 0, stream>>>(aggB, NJ * D);
  k_zero<<<1024, 256, 0, stream>>>(aggT, NT * D);
  k_scatter<<<(E1 * 16 + 255) / 256, 256, 0, stream>>>(htb, thj_src, thj_dst, aggA, E1);
  k_scatter<<<(E3 * 16 + 255) / 256, 256, 0, stream>>>(hjb, jkj_src, jkj_dst, aggB, E3);
  k_scatter<<<(E2 * 16 + 255) / 256, 256, 0, stream>>>(hjb, jht_src, jht_dst, aggT, E2);
  k_cvt_bf16<<<4096, 256, 0, stream>>>(aggA, aggAb, NJ * 16);
  k_cvt_bf16<<<4096, 256, 0, stream>>>(aggB, aggBb, NJ * 16);
  k_cvt_bf16<<<1024, 256, 0, stream>>>(aggT, aggTb, NT * 16);

  // -------- layer 1 fused GEMMs (WMMA bf16, async-staged A) --------
  k_gemm_tanh<4><<<NJ / 64, 256, 0, stream>>>(aggAb, hjb, aggBb, hjb, pw1,
                                              brel_thj1, brel_jkj1, hj1b);
  k_gemm_tanh<2><<<NT / 64, 256, 0, stream>>>(aggTb, htb, nullptr, nullptr, pwT,
                                              brel_jht1, nullptr, ht1b);

  // -------- layer 2 aggregations --------
  k_zero<<<4096, 256, 0, stream>>>(aggA, NJ * D);
  k_zero<<<4096, 256, 0, stream>>>(aggB, NJ * D);
  k_scatter<<<(E1 * 16 + 255) / 256, 256, 0, stream>>>(ht1b, thj_src, thj_dst, aggA, E1);
  k_scatter<<<(E3 * 16 + 255) / 256, 256, 0, stream>>>(hj1b, jkj_src, jkj_dst, aggB, E3);
  k_cvt_bf16<<<4096, 256, 0, stream>>>(aggA, aggAb, NJ * 16);
  k_cvt_bf16<<<4096, 256, 0, stream>>>(aggB, aggBb, NJ * 16);

  // -------- layer 2 fused GEMM --------
  k_gemm_tanh<4><<<NJ / 64, 256, 0, stream>>>(aggAb, hj1b, aggBb, hj1b, pw2,
                                              brel_thj2, brel_jkj2, hj2b);

  // -------- output heads --------
  float* loc = (float*)d_out;
  float* scl = loc + NJ;
  k_heads<<<NJ / 8, 256, 0, stream>>>(hj2b, W_out, b_out, loc, scl, NJ);
}
